// DenoisingNet_54185307407141
// MI455X (gfx1250) — compile-verified, matrix operand order checked by reading, not run on hardware
//
#include <hip/hip_runtime.h>

typedef __attribute__((ext_vector_type(16))) __bf16 v16bf;
typedef __attribute__((ext_vector_type(8)))  float  v8f;

#define D_IN  256   // feature dim
#define H_DIM 128   // hidden dim
// packed W fragments: 2 branches * 8 ntiles * 8 kblks * (32 lanes * 16 bf16)
#define PACKED_ELEMS (2 * 8 * 8 * 32 * 16)
#define PACKED_BYTES (PACKED_ELEMS * 2)

// ---------------------------------------------------------------------------
// Kernel 0: pack W_nb / W_self (row-major [256,128] f32) into bf16 WMMA
// B-fragment layout: frag(br,t,k): lane L holds column N = t*16 + (L&15),
// K = k*32 + 16*(L>>4) + j  for j in 0..15 (2 bf16 per VGPR).
// ---------------------------------------------------------------------------
__global__ void pack_w_kernel(const float* __restrict__ Wnb,
                              const float* __restrict__ Wself,
                              __bf16* __restrict__ packed) {
  int idx = blockIdx.x * blockDim.x + threadIdx.x;
  if (idx >= PACKED_ELEMS) return;
  int j    = idx & 15;
  int lane = (idx >> 4) & 31;
  int k    = (idx >> 9) & 7;
  int t    = (idx >> 12) & 7;
  int br   = (idx >> 15) & 1;
  int kk = k * 32 + ((lane >> 4) << 4) + j;   // K index in [0,256)
  int n  = t * 16 + (lane & 15);              // N index in [0,128)
  const float* W = br ? Wself : Wnb;
  packed[idx] = (__bf16)W[kk * H_DIM + n];
}

__global__ void zero_f32_kernel(float* __restrict__ p, int n) {
  int i = blockIdx.x * blockDim.x + threadIdx.x;
  if (i < n) p[i] = 0.0f;
}

// ---------------------------------------------------------------------------
// Kernel 1: per-node attention scores via WMMA.
// Packed W fragments are staged into LDS once per workgroup; each wave then
// computes a 16-node tile for BOTH branches with double-buffered ds feeds:
//   a_nb[n]   = sum_h relu(x[n]@W_nb  + b_nb)[h]  * W_att[h]
//   a_self[n] = sum_h relu(x[n]@W_self+ b_self)[h]* W_att[128+h]
// ---------------------------------------------------------------------------
__global__ __launch_bounds__(256) void node_scores_kernel(
    const float* __restrict__ x,
    const float* __restrict__ b_nb,
    const float* __restrict__ b_self,
    const float* __restrict__ W_att,
    const __bf16* __restrict__ packedW,
    float* __restrict__ a_nb,
    float* __restrict__ a_self,
    int N, int ntiles) {
  extern __shared__ __align__(16) char smem[];
  __bf16* lds_w = (__bf16*)smem;

  // cooperative stage of packed W into LDS (131072 B = 8192 x 16B)
  {
    const uint4* src = (const uint4*)packedW;
    uint4* dst = (uint4*)lds_w;
    for (int i = threadIdx.x; i < PACKED_BYTES / 16; i += blockDim.x)
      dst[i] = src[i];
  }
  __syncthreads();

  const int wave = threadIdx.x >> 5;
  const int lane = threadIdx.x & 31;
  const int tile = blockIdx.x * (blockDim.x >> 5) + wave;
  if (tile >= ntiles) return;            // wave-uniform: EXEC stays all-1s

  const int base  = tile * 16;
  const int khalf = lane >> 4;           // 0: K 0..15 of block, 1: K 16..31
  const int nlane = lane & 15;

  // A-fragment source row for this lane (16-bit A layout: lane = M + 16*Khalf)
  int arow = base + nlane;
  if (arow > N - 1) arow = N - 1;        // clamp (N=50000 is 16-aligned; safety)
  const float* __restrict__ xrow = x + (size_t)arow * D_IN + khalf * 16;

  // fragment fetch from LDS: frag(br,t,k) is 1KB; lane takes its 32B slice
  const v16bf* fbase = (const v16bf*)lds_w + lane;
#define LDFRAG(br, t, k) fbase[(size_t)((((br) << 3) + (t)) * 8 + (k)) * 32]

  v8f acc[2][8];
#pragma unroll
  for (int br = 0; br < 2; ++br)
#pragma unroll
    for (int t = 0; t < 8; ++t)
#pragma unroll
      for (int r = 0; r < 8; ++r) acc[br][t][r] = 0.0f;

  // prologue: A raw floats for k=0, first B fragment pair
  float araw[16];
#pragma unroll
  for (int j = 0; j < 16; ++j) araw[j] = xrow[j];
  v16bf afrag;
#pragma unroll
  for (int j = 0; j < 16; ++j) afrag[j] = (__bf16)araw[j];

  v16bf b0 = LDFRAG(0, 0, 0);
  v16bf b1 = LDFRAG(1, 0, 0);

#pragma unroll
  for (int k = 0; k < 8; ++k) {          // K blocks of 32
    // prefetch next K-block's A data (global) while WMMAs run
    float anext[16];
    if (k < 7) {
#pragma unroll
      for (int j = 0; j < 16; ++j) anext[j] = xrow[(k + 1) * 32 + j];
    }
#pragma unroll
    for (int t = 0; t < 8; ++t) {        // N tiles of 16
      // prefetch next fragment pair (double buffer)
      v16bf n0 = b0, n1 = b1;
      int tn = t + 1, kn = k;
      if (tn == 8) { tn = 0; kn = k + 1; }
      if (kn < 8) {
        n0 = LDFRAG(0, tn, kn);
        n1 = LDFRAG(1, tn, kn);
      }
      acc[0][t] = __builtin_amdgcn_wmma_f32_16x16x32_bf16(
          false, afrag, false, b0, (short)0, acc[0][t], false, false);
      acc[1][t] = __builtin_amdgcn_wmma_f32_16x16x32_bf16(
          false, afrag, false, b1, (short)0, acc[1][t], false, false);
      b0 = n0; b1 = n1;
    }
    if (k < 7) {
#pragma unroll
      for (int j = 0; j < 16; ++j) afrag[j] = (__bf16)anext[j];
    }
  }
#undef LDFRAG

  // bias + relu + W_att weight, reduce over N (16 lanes per half).
  // C layout: VGPR r, lanes 0-15 -> (M=r, N=lane); lanes 16-31 -> (M=r+8).
  float rowacc[2][8];
#pragma unroll
  for (int br = 0; br < 2; ++br)
#pragma unroll
    for (int r = 0; r < 8; ++r) rowacc[br][r] = 0.0f;

#pragma unroll
  for (int br = 0; br < 2; ++br) {
    const float* bias_p = br ? b_self : b_nb;
#pragma unroll
    for (int t = 0; t < 8; ++t) {
      float bias = bias_p[t * 16 + nlane];
      float watt = W_att[br * H_DIM + t * 16 + nlane];
#pragma unroll
      for (int r = 0; r < 8; ++r) {
        float c = acc[br][t][r] + bias;
        c = c > 0.0f ? c : 0.0f;
        float v = c * watt;
        v += __shfl_xor(v, 1, 32);       // butterfly within 16-lane halves
        v += __shfl_xor(v, 2, 32);
        v += __shfl_xor(v, 4, 32);
        v += __shfl_xor(v, 8, 32);
        rowacc[br][r] += v;
      }
    }
  }

  if (nlane == 0) {                      // lanes 0 (rows 0..7) and 16 (rows 8..15)
    int roff = khalf * 8;
#pragma unroll
    for (int r = 0; r < 8; ++r) {
      int node = base + roff + r;
      if (node < N) {
        a_nb[node]   = rowacc[0][r];
        a_self[node] = rowacc[1][r];
      }
    }
  }
}

// ---------------------------------------------------------------------------
// Kernel 2: hard-concrete gate per edge, masked values, rowsum scatter.
// ---------------------------------------------------------------------------
__global__ void edge_mask_kernel(const float* __restrict__ a_nb,
                                 const float* __restrict__ a_self,
                                 const float* __restrict__ b_att,
                                 const float* __restrict__ values,
                                 const float* __restrict__ noise,
                                 const int* __restrict__ row,
                                 const int* __restrict__ col,
                                 float* __restrict__ masked,
                                 float* __restrict__ rowsum, int E) {
  int e = blockIdx.x * blockDim.x + threadIdx.x;
  if (e >= E) return;
  int r = row[e], c = col[e];
  float log_alpha = a_nb[r] + a_self[c] + b_att[0];
  float u = noise[e] + 1e-7f;
  float z = __logf(u) - __logf(1.0f - u) + log_alpha;
  float gate = 1.0f / (1.0f + __expf(-z));
  float m = gate * 1.6f - 0.5f;          // ZETA - GAMMA = 1.6, GAMMA = -0.5
  m = fminf(fmaxf(m, 0.0f), 1.0f);
  float mv = values[e] * m;
  masked[e] = mv;
  atomicAdd(&rowsum[r], mv);
}

// ---------------------------------------------------------------------------
// Kernel 3: degree normalization factors, in place over rowsum.
// ---------------------------------------------------------------------------
__global__ void dinv_kernel(float* __restrict__ rowsum, int N) {
  int i = blockIdx.x * blockDim.x + threadIdx.x;
  if (i >= N) return;
  float r = rowsum[i] + 1e-10f;
  float d = (r > 0.0f) ? rsqrtf(r) : 0.0f;   // isinf(x^-0.5) -> 0
  rowsum[i] = d;
}

// ---------------------------------------------------------------------------
// Kernel 4: final edge values.
// ---------------------------------------------------------------------------
__global__ void finalize_kernel(const float* __restrict__ masked,
                                const float* __restrict__ dinv,
                                const int* __restrict__ row,
                                const int* __restrict__ col,
                                float* __restrict__ out, int E) {
  int e = blockIdx.x * blockDim.x + threadIdx.x;
  if (e >= E) return;
  out[e] = masked[e] * dinv[row[e]] * dinv[col[e]];
}

static inline size_t align256(size_t v) { return (v + 255) & ~(size_t)255; }

extern "C" void kernel_launch(void* const* d_in, const int* in_sizes, int n_in,
                              void* d_out, int out_size, void* d_ws, size_t ws_size,
                              hipStream_t stream) {
  // input order: x, W_nb, b_nb, W_self, b_self, W_att, b_att, values, noise, row, col
  const float* x      = (const float*)d_in[0];
  const float* W_nb   = (const float*)d_in[1];
  const float* b_nb   = (const float*)d_in[2];
  const float* W_self = (const float*)d_in[3];
  const float* b_self = (const float*)d_in[4];
  const float* W_att  = (const float*)d_in[5];
  const float* b_att  = (const float*)d_in[6];
  const float* values = (const float*)d_in[7];
  const float* noise  = (const float*)d_in[8];
  const int*   row    = (const int*)d_in[9];
  const int*   col    = (const int*)d_in[10];
  float* out = (float*)d_out;

  const int N = in_sizes[0] / D_IN;      // 50000
  const int E = in_sizes[7];             // 800000
  const int ntiles = (N + 15) / 16;      // 3125

  // workspace layout
  char* ws = (char*)d_ws;
  size_t off = 0;
  __bf16* packedW = (__bf16*)(ws + off); off += align256(PACKED_BYTES);
  float* a_nb     = (float*)(ws + off);  off += align256((size_t)N * sizeof(float));
  float* a_self   = (float*)(ws + off);  off += align256((size_t)N * sizeof(float));
  float* rowsum   = (float*)(ws + off);  off += align256((size_t)N * sizeof(float));
  float* masked   = (float*)(ws + off);  off += align256((size_t)E * sizeof(float));
  (void)ws_size; (void)n_in; (void)out_size;

  // allow 128KB dynamic LDS for the WMMA kernel (host-side attribute, not a
  // stream op -> safe under graph capture; idempotent across calls)
  static_assert(PACKED_BYTES == 131072, "packed W size");
  hipFuncSetAttribute((const void*)node_scores_kernel,
                      hipFuncAttributeMaxDynamicSharedMemorySize, PACKED_BYTES);

  pack_w_kernel<<<(PACKED_ELEMS + 255) / 256, 256, 0, stream>>>(W_nb, W_self, packedW);
  zero_f32_kernel<<<(N + 255) / 256, 256, 0, stream>>>(rowsum, N);
  node_scores_kernel<<<(ntiles + 7) / 8, 256, PACKED_BYTES, stream>>>(
      x, b_nb, b_self, W_att, packedW, a_nb, a_self, N, ntiles);
  edge_mask_kernel<<<(E + 255) / 256, 256, 0, stream>>>(
      a_nb, a_self, b_att, values, noise, row, col, masked, rowsum, E);
  dinv_kernel<<<(N + 255) / 256, 256, 0, stream>>>(rowsum, N);
  finalize_kernel<<<(E + 255) / 256, 256, 0, stream>>>(
      masked, rowsum, row, col, out, E);
}
